// MllamaTextCrossAttention_13228499272292
// MI455X (gfx1250) — matrix operand order repaired and checked by reading, not compiled, not run on hardware
//
#include <hip/hip_runtime.h>

#define HID 4096
#define NH 32
#define NKVH 8
#define HD 128
#define QL 2048
#define KVL 6404
#define KVP 6528            // KV padded to multiple of 128 (GEMM M-tile) and 32 (attn chunk)
#define RMS_EPS 1e-6f
#define SM_SCALE 0.08838834764831845f   // 1/sqrt(128)

typedef __attribute__((ext_vector_type(16))) __bf16 v16bf;
typedef __attribute__((ext_vector_type(8)))  float  v8f;

union Frag {
  v16bf v;
  uint4 q[2];
  unsigned short s[16];
};
union U4 { uint4 u; unsigned short s[8]; };

__device__ __forceinline__ unsigned short f2bf(float f) {
  union { float f; unsigned int u; } c; c.f = f;
  unsigned int u = c.u;
  u += 0x7fffu + ((u >> 16) & 1u);           // round-to-nearest-even
  return (unsigned short)(u >> 16);
}
__device__ __forceinline__ float bf2f(unsigned short h) {
  union { unsigned int u; float f; } c; c.u = ((unsigned int)h) << 16;
  return c.f;
}
__device__ __forceinline__ v8f zero8() {
  v8f z = {0.f,0.f,0.f,0.f,0.f,0.f,0.f,0.f};
  return z;
}
__device__ __forceinline__ v8f wmma_bf16(const Frag& a, const Frag& b, v8f c) {
  return __builtin_amdgcn_wmma_f32_16x16x32_bf16(false, a.v, false, b.v,
                                                 (short)0, c, false, false);
}

// ---------------- fp32 -> bf16 converts ----------------
__global__ void k_cvt(const float* __restrict__ s, unsigned short* __restrict__ d, long n) {
  long stride = (long)gridDim.x * blockDim.x;
  for (long i = (long)blockIdx.x * blockDim.x + threadIdx.x; i < n; i += stride)
    d[i] = f2bf(s[i]);
}

__global__ void k_cvt_pad(const float* __restrict__ s, unsigned short* __restrict__ d,
                          int rows_src, int rows_dst, int w) {
  long n = (long)rows_dst * w;
  long stride = (long)gridDim.x * blockDim.x;
  for (long i = (long)blockIdx.x * blockDim.x + threadIdx.x; i < n; i += stride) {
    long row = i / w;
    d[i] = (row < rows_src) ? f2bf(s[i]) : (unsigned short)0;
  }
}

// ---------------- bf16 GEMM: C[m,n] = sum_k A[m,k] * B[n,k] ----------------
// block = 256 threads = 8 waves arranged 2(M) x 4(N); block tile 128(M) x 256(N);
// wave tile 64x64 = 16 WMMA per 16 b128 loads per k-step (4x fragment reuse).
// STORE: 0 = bf16 row-major, 1 = bf16 transposed (C[n*ldc + m]), 2 = f32 row-major
template<int STORE>
__global__ __launch_bounds__(256)
void k_gemm(const unsigned short* __restrict__ A,
            const unsigned short* __restrict__ B,
            void* __restrict__ Cv,
            int M, int N, int K, int lda, int ldb, int ldc)
{
  const int lane = threadIdx.x & 31;
  const int wave = threadIdx.x >> 5;
  const int row0 = blockIdx.y * 128 + (wave >> 2) * 64;
  const int col0 = blockIdx.x * 256 + (wave & 3) * 64;
  const int mA = lane & 15;
  const int kg = lane >> 4;
  const int nB = lane & 15;

  const unsigned short* arow[4];
  const unsigned short* brow[4];
#pragma unroll
  for (int i = 0; i < 4; i++) {
    arow[i] = A + (size_t)(row0 + i*16 + mA) * lda;
    brow[i] = B + (size_t)(col0 + i*16 + nB) * ldb;
  }

  v8f acc[4][4];
#pragma unroll
  for (int i = 0; i < 4; i++)
#pragma unroll
    for (int j = 0; j < 4; j++) acc[i][j] = zero8();

  for (int k0 = 0; k0 < K; k0 += 32) {
    Frag fa[4], fb[4];
    const int ak = k0 + 8 * kg;    // A frag: K runs {8kg..8kg+7} and {8kg+16..8kg+23}
    const int bk = k0 + 8 * kg;    // B frag: K runs {8kg..8kg+7} and {8kg+16..8kg+23}
#pragma unroll
    for (int i = 0; i < 4; i++) {
      fa[i].q[0] = *(const uint4*)(arow[i] + ak);
      fa[i].q[1] = *(const uint4*)(arow[i] + ak + 16);
      fb[i].q[0] = *(const uint4*)(brow[i] + bk);
      fb[i].q[1] = *(const uint4*)(brow[i] + bk + 16);
    }
    __builtin_prefetch(arow[0] + ak + 64, 0, 3);
    __builtin_prefetch(brow[0] + bk + 64, 0, 3);
#pragma unroll
    for (int i = 0; i < 4; i++)
#pragma unroll
      for (int j = 0; j < 4; j++)
        acc[i][j] = wmma_bf16(fa[i], fb[j], acc[i][j]);
  }

  const int mh = kg * 8;
  if (STORE == 2) {
    float* C = (float*)Cv;
#pragma unroll
    for (int tm = 0; tm < 4; tm++)
#pragma unroll
      for (int tn = 0; tn < 4; tn++)
#pragma unroll
        for (int r = 0; r < 8; r++)
          C[(size_t)(row0 + tm*16 + mh + r) * ldc + (col0 + tn*16 + nB)] = acc[tm][tn][r];
  } else if (STORE == 0) {
    unsigned short* C = (unsigned short*)Cv;
#pragma unroll
    for (int tm = 0; tm < 4; tm++)
#pragma unroll
      for (int tn = 0; tn < 4; tn++)
#pragma unroll
        for (int r = 0; r < 8; r++)
          C[(size_t)(row0 + tm*16 + mh + r) * ldc + (col0 + tn*16 + nB)] = f2bf(acc[tm][tn][r]);
  } else {
    unsigned short* C = (unsigned short*)Cv;
#pragma unroll
    for (int tm = 0; tm < 4; tm++)
#pragma unroll
      for (int tn = 0; tn < 4; tn++) {
        U4 pack;
#pragma unroll
        for (int r = 0; r < 8; r++) pack.s[r] = f2bf(acc[tm][tn][r]);
        *(uint4*)(C + (size_t)(col0 + tn*16 + nB) * ldc + (row0 + tm*16 + mh)) = pack.u;
      }
  }
}

// ---------------- per-128 RMSNorm (in-place on bf16), one wave per group ----------------
__global__ __launch_bounds__(256)
void k_rmsnorm(unsigned short* __restrict__ x, const float* __restrict__ w, long ngroups)
{
  const long g = (long)blockIdx.x * 8 + (threadIdx.x >> 5);
  if (g >= ngroups) return;
  const int lane = threadIdx.x & 31;
  unsigned short* p = x + g * HD;
  float v[4];
  float ss = 0.f;
  for (int i = 0; i < 4; i++) {
    v[i] = bf2f(p[lane + 32*i]);
    ss += v[i] * v[i];
  }
  for (int d = 16; d >= 1; d >>= 1) ss += __shfl_xor(ss, d, 32);
  const float r = rsqrtf(ss * (1.0f / HD) + RMS_EPS);
  for (int i = 0; i < 4; i++)
    p[lane + 32*i] = f2bf(w[lane + 32*i] * (v[i] * r));
}

// ---------------- flash attention: one wave per (head, 16-row q tile) ----------------
__global__ __launch_bounds__(256)
void k_attn(const unsigned short* __restrict__ qb,   // [QL, NH*HD]
            const unsigned short* __restrict__ kb,   // [KVP, NKVH*HD]
            const unsigned short* __restrict__ vt,   // [NKVH*HD, KVP] (V transposed)
            unsigned short* __restrict__ ob)         // [QL, NH*HD]
{
  __shared__ unsigned short psh[8][16 * 32];
  const int lane = threadIdx.x & 31;
  const int wave = threadIdx.x >> 5;
  const int task = blockIdx.x * 8 + wave;   // 32 heads * 128 q-tiles = 4096 tasks
  const int h   = task >> 7;
  const int qt  = task & 127;
  const int kvh = h >> 2;                   // GQA: 4 q-heads per kv-head
  const int q0  = qt * 16;

  const int mA = lane & 15;
  const int kg = lane >> 4;
  const int nB = lane & 15;
  const int mh = kg * 8;

  // resident Q A-fragments over d = 0..127 (4 chunks of 32)
  Frag aq[4];
  {
    const unsigned short* qp = qb + (size_t)(q0 + mA) * (NH * HD) + h * HD;
    for (int dc = 0; dc < 4; dc++) {
      const int ak = dc * 32 + 8 * kg;
      aq[dc].q[0] = *(const uint4*)(qp + ak);
      aq[dc].q[1] = *(const uint4*)(qp + ak + 16);
    }
  }

  v8f o[8];
  for (int i = 0; i < 8; i++) o[i] = zero8();
  float mr[8], lr[8];
  for (int r = 0; r < 8; r++) { mr[r] = -1e30f; lr[r] = 0.f; }

  const unsigned short* kpb = kb + kvh * HD;
  const unsigned short* vpb = vt + (size_t)kvh * HD * KVP;

  for (int kv0 = 0; kv0 < KVP; kv0 += 32) {
    // ---- S = Q K^T for a 16x32 chunk of columns (two 16x16 C tiles) ----
    float sv[2][8];
    float cm[8];
    for (int r = 0; r < 8; r++) cm[r] = -1e30f;
    for (int t = 0; t < 2; t++) {
      v8f s = zero8();
      const unsigned short* kr = kpb + (size_t)(kv0 + t*16 + nB) * (NKVH * HD);
      for (int dc = 0; dc < 4; dc++) {
        Frag fb;
        const int bk = dc * 32 + 8 * kg;
        fb.q[0] = *(const uint4*)(kr + bk);
        fb.q[1] = *(const uint4*)(kr + bk + 16);
        s = wmma_bf16(aq[dc], fb, s);
      }
      const bool masked = (kv0 + t*16 + nB) >= KVL;
      for (int r = 0; r < 8; r++) {
        float x = masked ? -1e30f : s[r] * SM_SCALE;
        sv[t][r] = x;
        cm[r] = fmaxf(cm[r], x);
      }
    }
    // row-max across the 16 lanes holding this row's columns
    for (int d = 8; d >= 1; d >>= 1)
      for (int r = 0; r < 8; r++)
        cm[r] = fmaxf(cm[r], __shfl_xor(cm[r], d, 32));

    float resc[8], rs[8];
    for (int r = 0; r < 8; r++) {
      float mn = fmaxf(mr[r], cm[r]);
      resc[r] = __expf(mr[r] - mn);
      mr[r] = mn;
      rs[r] = 0.f;
    }

    // exp + stage P (16x32, row-major bf16) into this wave's LDS slab
    unsigned short* myp = psh[wave];
    for (int t = 0; t < 2; t++)
      for (int r = 0; r < 8; r++) {
        float e = __expf(sv[t][r] - mr[r]);
        rs[r] += e;
        myp[(mh + r) * 32 + t*16 + nB] = f2bf(e);
      }
    for (int d = 8; d >= 1; d >>= 1)
      for (int r = 0; r < 8; r++)
        rs[r] += __shfl_xor(rs[r], d, 32);
    for (int r = 0; r < 8; r++) lr[r] = lr[r] * resc[r] + rs[r];
    for (int dt = 0; dt < 8; dt++)
      for (int r = 0; r < 8; r++)
        o[dt][r] *= resc[r];

    __syncthreads();  // uniform: every wave runs identical trip count
    Frag fp;          // reload P as an A-fragment
    for (int j = 0; j < 8; j++) fp.s[j]     = myp[mA * 32 + 8*kg + j];
    for (int j = 0; j < 8; j++) fp.s[8 + j] = myp[mA * 32 + 8*kg + 16 + j];
    __syncthreads();

    // ---- O += P V : B-fragments from V^T, K contiguous along kv ----
    for (int dt = 0; dt < 8; dt++) {
      Frag fv;
      const unsigned short* vr = vpb + (size_t)(dt*16 + nB) * KVP + kv0 + 8 * kg;
      fv.q[0] = *(const uint4*)(vr);
      fv.q[1] = *(const uint4*)(vr + 16);
      o[dt] = wmma_bf16(fp, fv, o[dt]);
    }
  }

  // epilogue: normalize and store bf16 [QL, NH*HD]
  unsigned short* op = ob + (size_t)(q0 + mh) * (NH * HD) + h * HD;
  for (int dt = 0; dt < 8; dt++)
    for (int r = 0; r < 8; r++)
      op[(size_t)r * (NH * HD) + dt*16 + nB] = f2bf(o[dt][r] / lr[r]);
}

// ---------------- launch ----------------
extern "C" void kernel_launch(void* const* d_in, const int* in_sizes, int n_in,
                              void* d_out, int out_size, void* d_ws, size_t ws_size,
                              hipStream_t stream)
{
  const float* hidden = (const float*)d_in[0];
  const float* cross  = (const float*)d_in[1];
  const float* Wq = (const float*)d_in[2];
  const float* Wk = (const float*)d_in[3];
  const float* Wv = (const float*)d_in[4];
  const float* Wo = (const float*)d_in[5];
  const float* qw = (const float*)d_in[6];
  const float* kw = (const float*)d_in[7];
  float* out = (float*)d_out;

  char* ws = (char*)d_ws;
  size_t off = 0;
  auto take = [&](size_t elems) -> unsigned short* {
    char* p = ws + off;
    off += (elems * 2 + 255) & ~(size_t)255;
    return (unsigned short*)p;
  };
  unsigned short* hsb = take((size_t)QL * HID);          // hidden bf16
  unsigned short* csb = take((size_t)KVP * HID);         // cross bf16 (padded)
  unsigned short* wqb = take((size_t)HID * HID);
  unsigned short* wkb = take((size_t)NKVH * HD * HID);
  unsigned short* wvb = take((size_t)NKVH * HD * HID);
  unsigned short* wob = take((size_t)HID * HID);
  unsigned short* qbf = take((size_t)QL * HID);          // q projected (then normed)
  unsigned short* kbf = take((size_t)KVP * NKVH * HD);   // k projected (then normed)
  unsigned short* vtb = take((size_t)NKVH * HD * KVP);   // v projected, transposed
  unsigned short* aob = take((size_t)QL * HID);          // attention output
  (void)ws_size; (void)in_sizes; (void)n_in; (void)out_size;

  // 1) fp32 -> bf16
  k_cvt    <<<4096, 256, 0, stream>>>(hidden, hsb, (long)QL * HID);
  k_cvt_pad<<<4096, 256, 0, stream>>>(cross, csb, KVL, KVP, HID);
  k_cvt    <<<4096, 256, 0, stream>>>(Wq, wqb, (long)HID * HID);
  k_cvt    <<<2048, 256, 0, stream>>>(Wk, wkb, (long)NKVH * HD * HID);
  k_cvt    <<<2048, 256, 0, stream>>>(Wv, wvb, (long)NKVH * HD * HID);
  k_cvt    <<<4096, 256, 0, stream>>>(Wo, wob, (long)HID * HID);

  // 2) projections (y = x @ W^T); block tile 128(M) x 256(N)
  dim3 blk(256);
  k_gemm<0><<<dim3(HID/256, QL/128), blk, 0, stream>>>(hsb, wqb, qbf,
      QL, HID, HID, HID, HID, HID);
  k_gemm<0><<<dim3((NKVH*HD)/256, KVP/128), blk, 0, stream>>>(csb, wkb, kbf,
      KVP, NKVH*HD, HID, HID, HID, NKVH*HD);
  k_gemm<1><<<dim3((NKVH*HD)/256, KVP/128), blk, 0, stream>>>(csb, wvb, vtb,
      KVP, NKVH*HD, HID, HID, HID, KVP);

  // 3) RMSNorm q (per head-128) and k (per kv-head-128)
  k_rmsnorm<<<(QL * NH) / 8, 256, 0, stream>>>(qbf, qw, (long)QL * NH);
  k_rmsnorm<<<(KVP * NKVH) / 8, 256, 0, stream>>>(kbf, kw, (long)KVP * NKVH);

  // 4) flash attention (32 heads * 128 q-tiles = 4096 wave tasks, 8 waves/block)
  k_attn<<<(NH * (QL/16)) / 8, 256, 0, stream>>>(qbf, kbf, vtb, aob);

  // 5) output projection -> fp32 d_out
  k_gemm<2><<<dim3(HID/256, QL/128), blk, 0, stream>>>(aob, wob, out,
      QL, HID, HID, HID, HID, HID);
}
